// WindowAttention_8048768713330
// MI455X (gfx1250) — matrix operand order repaired and checked by reading, not compiled, not run on hardware
//
#include <hip/hip_runtime.h>

typedef unsigned short u16;
typedef unsigned int   u32;
typedef unsigned long long u64;

typedef __bf16 bf16;
typedef bf16  v16bf __attribute__((ext_vector_type(16)));
typedef float v8f   __attribute__((ext_vector_type(8)));

#define BATCH 2048
#define SEQ   64
#define EMBD  512
#define HEADS 16
#define HDIM  32
#define C3    1536

// ---------- workspace layout (bytes) ----------
#define OFF_Q    ((size_t)0)
#define OFF_K    (OFF_Q  + (size_t)BATCH*HEADS*SEQ*HDIM*2)   // 128 MB each
#define OFF_VT   (OFF_K  + (size_t)BATCH*HEADS*SEQ*HDIM*2)
#define OFF_AOUT (OFF_VT + (size_t)BATCH*HEADS*SEQ*HDIM*2)
#define OFF_XB   OFF_AOUT  /* bf16 x aliases aout: x consumed by qkv before attn writes */
#define OFF_WQ   (OFF_AOUT + (size_t)BATCH*SEQ*EMBD*2)
#define OFF_WP   (OFF_WQ + (size_t)C3*EMBD*2)
#define OFF_BIAS (OFF_WP + (size_t)EMBD*EMBD*2)
#define OFF_TAB  (OFF_BIAS + (size_t)HEADS*SEQ*SEQ*4)

// ---------- helpers ----------
__device__ __forceinline__ u16 f2bf(float f) {
  u32 u = __builtin_bit_cast(u32, f);
  u32 r = (u + 0x7FFFu + ((u >> 16) & 1u)) >> 16;   // RNE
  return (u16)r;
}
__device__ __forceinline__ u32 pk2(float lo, float hi) {
  return (u32)f2bf(lo) | ((u32)f2bf(hi) << 16);
}
__device__ __forceinline__ float bf2f(u16 h) {
  return __builtin_bit_cast(float, (u32)h << 16);
}

union FragBf { v16bf v; uint4 u4[2]; };

__device__ __forceinline__ v8f wmma_bf16(v16bf a, v16bf b, v8f c) {
  // D = A(16x32 bf16) * B(32x16 bf16) + C(16x16 f32)
  return __builtin_amdgcn_wmma_f32_16x16x32_bf16(false, a, false, b,
                                                 (short)0, c, false, false);
}
__device__ __forceinline__ v8f vzero8() {
  v8f z = {0.f,0.f,0.f,0.f,0.f,0.f,0.f,0.f};
  return z;
}
__device__ __forceinline__ void unpack16(uint4 a, uint4 b, float* o) {
  const u32 u[8] = {a.x,a.y,a.z,a.w,b.x,b.y,b.z,b.w};
#pragma unroll
  for (int i = 0; i < 8; ++i) {
    o[2*i]   = bf2f((u16)(u[i] & 0xFFFFu));
    o[2*i+1] = bf2f((u16)(u[i] >> 16));
  }
}
// 32-byte global -> LDS async copy (2x B128, ASYNCcnt += 2 per wave)
__device__ __forceinline__ void async_cp32(u32 lds_addr, const u16* g) {
  const u64 ga = (u64)(size_t)g;
  asm volatile("global_load_async_to_lds_b128 %0, %1, off"
               :: "v"(lds_addr), "v"(ga) : "memory");
  asm volatile("global_load_async_to_lds_b128 %0, %1, off offset:16"
               :: "v"(lds_addr), "v"(ga) : "memory");
}
__device__ __forceinline__ u32 lds_off(const void* p) {
  return (u32)(size_t)p;   // LDS aperture lives in addr[63:32]; low 32 = LDS offset
}

// ---------- kernel 0: fp32 -> bf16 bulk convert (8 elts/thread) ----------
__global__ __launch_bounds__(256)
void cvt_bf16_kernel(const float* __restrict__ src, u16* __restrict__ dst) {
  const size_t i = ((size_t)blockIdx.x * 256 + threadIdx.x) * 8;
  const float4 a = ((const float4*)(src + i))[0];
  const float4 b = ((const float4*)(src + i))[1];
  uint4 o;
  o.x = pk2(a.x, a.y); o.y = pk2(a.z, a.w);
  o.z = pk2(b.x, b.y); o.w = pk2(b.z, b.w);
  *(uint4*)(dst + i) = o;
}

// ---------- kernel 1: CPB MLP table (225 x 16) ----------
__global__ __launch_bounds__(128)
void cpb_tab_kernel(const float* __restrict__ rel_table,
                    const float* __restrict__ w1, const float* __restrict__ b1,
                    const float* __restrict__ w2, float* __restrict__ tab) {
  __shared__ float part[128][16];
  const int i = blockIdx.x;        // 0..224
  const int t = threadIdx.x;       // 0..127
  const float t0 = rel_table[i*2+0], t1 = rel_table[i*2+1];
  float acc[16];
#pragma unroll
  for (int hh = 0; hh < 16; ++hh) acc[hh] = 0.f;
  for (int j = t; j < 512; j += 128) {
    float hv = fmaxf(w1[j*2]*t0 + w1[j*2+1]*t1 + b1[j], 0.f);
#pragma unroll
    for (int hh = 0; hh < 16; ++hh) acc[hh] += hv * w2[hh*512 + j];
  }
#pragma unroll
  for (int hh = 0; hh < 16; ++hh) part[t][hh] = acc[hh];
  __syncthreads();
  if (t < 16) {
    float s = 0.f;
    for (int j = 0; j < 128; ++j) s += part[j][t];
    tab[i*16 + t] = s;
  }
}

// ---------- kernel 2: gather + 16*sigmoid -> bias[H][64][64] ----------
__global__ __launch_bounds__(256)
void cpb_bias_kernel(const float* __restrict__ tab, const int* __restrict__ rel_index,
                     float* __restrict__ biasb) {
  const int idx = blockIdx.x * 256 + threadIdx.x;   // < 16*64*64
  const int h  = idx >> 12;
  const int nm = idx & 4095;
  const float x = tab[rel_index[nm]*16 + h];
  biasb[h*4096 + nm] = 16.f / (1.f + __expf(-x));
}

// ---------- kernel 3: QKV GEMM, bf16 WMMA, double-buffered async staging ----
// C(131072 x 1536) = Xb(131072x512,bf16) * Wqb^T ; tile 128x128, K-step 32.
__global__ __launch_bounds__(256)
void qkv_gemm_kernel(const u16* __restrict__ Xb, const u16* __restrict__ Wb,
                     const float* __restrict__ qb, const float* __restrict__ vb,
                     u16* __restrict__ qg, u16* __restrict__ kg, u16* __restrict__ vtg) {
  __shared__ __align__(16) u16 sA[2][128*40];   // 32 halfs + 8 pad per row (80B stride)
  __shared__ __align__(16) u16 sB[2][128*40];
  const int t    = threadIdx.x;
  const int m0   = blockIdx.y * 128;
  const int n0   = blockIdx.x * 128;
  const int lane = t & 31, w = t >> 5;
  const int wm = w >> 2, wn = w & 3;            // 2x4 wave grid, each 64x32
  const int lm = lane & 15, lh = lane >> 4;
  const int srow = t >> 1;
  const int sseg = (t & 1) * 16;

  const u32 ldsA0 = lds_off(&sA[0][srow*40 + sseg]);
  const u32 ldsA1 = lds_off(&sA[1][srow*40 + sseg]);
  const u32 ldsB0 = lds_off(&sB[0][srow*40 + sseg]);
  const u32 ldsB1 = lds_off(&sB[1][srow*40 + sseg]);
  const u16* gA = Xb + (size_t)(m0 + srow) * EMBD + sseg;
  const u16* gB = Wb + (size_t)(n0 + srow) * EMBD + sseg;

  v8f acc[4][2];
#pragma unroll
  for (int i = 0; i < 4; ++i)
#pragma unroll
    for (int j = 0; j < 2; ++j) acc[i][j] = vzero8();

  // prime buffer 0
  async_cp32(ldsA0, gA);
  async_cp32(ldsB0, gB);

  for (int ks = 0; ks < 16; ++ks) {
    const int buf = ks & 1;
    if (ks < 15) {   // prefetch next K tile into the other buffer
      const int kn = (ks + 1) * 32;
      async_cp32(buf ? ldsA0 : ldsA1, gA + kn);
      async_cp32(buf ? ldsB0 : ldsB1, gB + kn);
      asm volatile("s_wait_asynccnt 0x4" ::: "memory");  // current tile done
    } else {
      asm volatile("s_wait_asynccnt 0x0" ::: "memory");
    }
    __syncthreads();

    FragBf a[4], bb[2];
#pragma unroll
    for (int mt = 0; mt < 4; ++mt) {
      const uint4* p = (const uint4*)&sA[buf][(wm*64 + mt*16 + lm)*40];
      a[mt].u4[0] = p[2*lh];        // K in [16*lh, 16*lh+8)
      a[mt].u4[1] = p[2*lh + 1];    // K in [16*lh+8, 16*lh+16)
    }
#pragma unroll
    for (int nt = 0; nt < 2; ++nt) {
      const uint4* p = (const uint4*)&sB[buf][(wn*32 + nt*16 + lm)*40];
      bb[nt].u4[0] = p[2*lh];       // K in [16*lh, 16*lh+16)
      bb[nt].u4[1] = p[2*lh + 1];
    }
#pragma unroll
    for (int mt = 0; mt < 4; ++mt)
#pragma unroll
      for (int nt = 0; nt < 2; ++nt)
        acc[mt][nt] = wmma_bf16(a[mt].v, bb[nt].v, acc[mt][nt]);
    __syncthreads();
  }

  // epilogue: add q/v bias, split into q, k, v^T (bf16)
#pragma unroll
  for (int mt = 0; mt < 4; ++mt) {
#pragma unroll
    for (int nt = 0; nt < 2; ++nt) {
#pragma unroll
      for (int r = 0; r < 8; ++r) {
        float vv = acc[mt][nt][r];
        const int R  = m0 + wm*64 + mt*16 + r + 8*lh;
        const int Cc = n0 + wn*32 + nt*16 + lm;
        const int which = Cc >> 9;
        const int inner = Cc & 511;
        const float bias = (which == 0) ? qb[inner] : ((which == 2) ? vb[inner] : 0.f);
        const u16 bvv = f2bf(vv + bias);
        const int h = inner >> 5, d = inner & 31, bwin = R >> 6, n = R & 63;
        const size_t head = (size_t)(bwin*HEADS + h);
        if (which == 0)      qg[(head*SEQ + n)*HDIM + d]  = bvv;
        else if (which == 1) kg[(head*SEQ + n)*HDIM + d]  = bvv;
        else                 vtg[(head*HDIM + d)*SEQ + n] = bvv;
      }
    }
  }
}

// ---------- kernel 4: per-(window,head) cosine attention (WMMA) ----------
__global__ __launch_bounds__(128)
void attn_kernel(const u16* __restrict__ qg, const u16* __restrict__ kg,
                 const u16* __restrict__ vtg, const float* __restrict__ biasb,
                 const float* __restrict__ flex, u16* __restrict__ aout) {
  __shared__ __align__(16) u16 qn_s[64*40];   // 64x32, 80B row stride
  __shared__ __align__(16) u16 kn_s[64*40];
  __shared__ __align__(16) u16 vt_s[32*72];   // v^T 32x64, 144B row stride
  __shared__ __align__(16) u16 p_s[64*72];    // softmax probs 64x64
  const int t = threadIdx.x;                  // 128 threads = 4 waves
  const int bwin = blockIdx.x >> 4;
  const int h    = blockIdx.x & 15;
  const int lane = t & 31, w = t >> 5;
  const int lm = lane & 15, lh = lane >> 4;
  const size_t head = (size_t)(bwin*HEADS + h);

  // async-copy v^T tile while we normalize q,k in VALU
  {
    const int dr = t >> 2, cs = (t & 3) * 16;
    async_cp32(lds_off(&vt_s[dr*72 + cs]), vtg + (head*HDIM + dr)*SEQ + cs);
  }
  // load + L2-normalize q,k rows (2 threads per row)
  {
    const int row = t >> 1, seg = (t & 1) * 16;
    const uint4* qp = (const uint4*)(qg + (head*SEQ + row)*HDIM + seg);
    const uint4* kp = (const uint4*)(kg + (head*SEQ + row)*HDIM + seg);
    const uint4 qu0 = qp[0], qu1 = qp[1];
    const uint4 ku0 = kp[0], ku1 = kp[1];
    float qv[16], kv[16];
    unpack16(qu0, qu1, qv);
    unpack16(ku0, ku1, kv);
    float qs = 0.f, ks = 0.f;
#pragma unroll
    for (int i = 0; i < 16; ++i) { qs += qv[i]*qv[i]; ks += kv[i]*kv[i]; }
    qs += __shfl_xor(qs, 1);
    ks += __shfl_xor(ks, 1);
    const float qi = 1.f / fmaxf(sqrtf(qs), 1e-12f);
    const float ki = 1.f / fmaxf(sqrtf(ks), 1e-12f);
    u32* qd = (u32*)&qn_s[row*40 + seg];
    u32* kd = (u32*)&kn_s[row*40 + seg];
#pragma unroll
    for (int i = 0; i < 8; ++i) {
      qd[i] = pk2(qv[2*i]*qi, qv[2*i+1]*qi);
      kd[i] = pk2(kv[2*i]*ki, kv[2*i+1]*ki);
    }
  }
  asm volatile("s_wait_asynccnt 0x0" ::: "memory");
  __syncthreads();

  const float sc = __expf(fminf(flex[h], 4.6051702f));  // clamp at ln(100)

  // S = qn @ kn^T : wave w owns query rows [16w,16w+16), all 64 key cols
  v8f sfr[4];
#pragma unroll
  for (int nt = 0; nt < 4; ++nt) sfr[nt] = vzero8();
  FragBf af;
  {
    const uint4* pa = (const uint4*)&qn_s[(w*16 + lm)*40];
    af.u4[0] = pa[2*lh];
    af.u4[1] = pa[2*lh + 1];
  }
#pragma unroll
  for (int nt = 0; nt < 4; ++nt) {
    FragBf bb;
    const uint4* pb2 = (const uint4*)&kn_s[(nt*16 + lm)*40];
    bb.u4[0] = pb2[2*lh];
    bb.u4[1] = pb2[2*lh + 1];
    sfr[nt] = wmma_bf16(af.v, bb.v, sfr[nt]);
  }

  // scale + CPB bias + row softmax (rows live in 16-lane halves; xor-reduce)
  const float* bp = biasb + (size_t)h * 4096;
#pragma unroll
  for (int r = 0; r < 8; ++r) {
    const int row = w*16 + r + 8*lh;
    float vv[4];
#pragma unroll
    for (int nt = 0; nt < 4; ++nt)
      vv[nt] = sfr[nt][r]*sc + bp[row*64 + nt*16 + lm];
    float mx = fmaxf(fmaxf(vv[0], vv[1]), fmaxf(vv[2], vv[3]));
    mx = fmaxf(mx, __shfl_xor(mx, 1));
    mx = fmaxf(mx, __shfl_xor(mx, 2));
    mx = fmaxf(mx, __shfl_xor(mx, 4));
    mx = fmaxf(mx, __shfl_xor(mx, 8));
    float sum = 0.f;
#pragma unroll
    for (int nt = 0; nt < 4; ++nt) { vv[nt] = __expf(vv[nt] - mx); sum += vv[nt]; }
    sum += __shfl_xor(sum, 1);
    sum += __shfl_xor(sum, 2);
    sum += __shfl_xor(sum, 4);
    sum += __shfl_xor(sum, 8);
    const float inv = 1.f / sum;
#pragma unroll
    for (int nt = 0; nt < 4; ++nt)
      p_s[row*72 + nt*16 + lm] = f2bf(vv[nt]*inv);
  }
  __syncthreads();

  // O = P @ V : A = P rows, B^T = v^T rows; K=64 -> two 32-steps
  v8f ofr[2];
#pragma unroll
  for (int nt = 0; nt < 2; ++nt) ofr[nt] = vzero8();
#pragma unroll
  for (int ks = 0; ks < 2; ++ks) {
    FragBf af2;
    const uint4* pp = (const uint4*)&p_s[(w*16 + lm)*72];
    af2.u4[0] = pp[ks*4 + 2*lh];
    af2.u4[1] = pp[ks*4 + 2*lh + 1];
#pragma unroll
    for (int nt = 0; nt < 2; ++nt) {
      FragBf bb2;
      const uint4* pv = (const uint4*)&vt_s[(nt*16 + lm)*72];
      bb2.u4[0] = pv[ks*4 + 2*lh];
      bb2.u4[1] = pv[ks*4 + 2*lh + 1];
      ofr[nt] = wmma_bf16(af2.v, bb2.v, ofr[nt]);
    }
  }
#pragma unroll
  for (int nt = 0; nt < 2; ++nt)
#pragma unroll
    for (int r = 0; r < 8; ++r) {
      const int n = w*16 + r + 8*lh, d = nt*16 + lm;
      aout[((size_t)(bwin*SEQ + n))*EMBD + h*HDIM + d] = f2bf(ofr[nt][r]);
    }
}

// ---------- kernel 5: output projection GEMM (bf16 WMMA, fp32 out) ----------
__global__ __launch_bounds__(256)
void proj_gemm_kernel(const u16* __restrict__ A, const u16* __restrict__ Wb,
                      const float* __restrict__ pb, float* __restrict__ out) {
  __shared__ __align__(16) u16 sA[2][128*40];
  __shared__ __align__(16) u16 sB[2][128*40];
  const int t    = threadIdx.x;
  const int m0   = blockIdx.y * 128;
  const int n0   = blockIdx.x * 128;
  const int lane = t & 31, w = t >> 5;
  const int wm = w >> 2, wn = w & 3;
  const int lm = lane & 15, lh = lane >> 4;
  const int srow = t >> 1;
  const int sseg = (t & 1) * 16;

  const u32 ldsA0 = lds_off(&sA[0][srow*40 + sseg]);
  const u32 ldsA1 = lds_off(&sA[1][srow*40 + sseg]);
  const u32 ldsB0 = lds_off(&sB[0][srow*40 + sseg]);
  const u32 ldsB1 = lds_off(&sB[1][srow*40 + sseg]);
  const u16* gA = A  + (size_t)(m0 + srow) * EMBD + sseg;
  const u16* gB = Wb + (size_t)(n0 + srow) * EMBD + sseg;

  v8f acc[4][2];
#pragma unroll
  for (int i = 0; i < 4; ++i)
#pragma unroll
    for (int j = 0; j < 2; ++j) acc[i][j] = vzero8();

  async_cp32(ldsA0, gA);
  async_cp32(ldsB0, gB);

  for (int ks = 0; ks < 16; ++ks) {
    const int buf = ks & 1;
    if (ks < 15) {
      const int kn = (ks + 1) * 32;
      async_cp32(buf ? ldsA0 : ldsA1, gA + kn);
      async_cp32(buf ? ldsB0 : ldsB1, gB + kn);
      asm volatile("s_wait_asynccnt 0x4" ::: "memory");
    } else {
      asm volatile("s_wait_asynccnt 0x0" ::: "memory");
    }
    __syncthreads();

    FragBf a[4], bb[2];
#pragma unroll
    for (int mt = 0; mt < 4; ++mt) {
      const uint4* p = (const uint4*)&sA[buf][(wm*64 + mt*16 + lm)*40];
      a[mt].u4[0] = p[2*lh];
      a[mt].u4[1] = p[2*lh + 1];
    }
#pragma unroll
    for (int nt = 0; nt < 2; ++nt) {
      const uint4* p = (const uint4*)&sB[buf][(wn*32 + nt*16 + lm)*40];
      bb[nt].u4[0] = p[2*lh];
      bb[nt].u4[1] = p[2*lh + 1];
    }
#pragma unroll
    for (int mt = 0; mt < 4; ++mt)
#pragma unroll
      for (int nt = 0; nt < 2; ++nt)
        acc[mt][nt] = wmma_bf16(a[mt].v, bb[nt].v, acc[mt][nt]);
    __syncthreads();
  }

#pragma unroll
  for (int mt = 0; mt < 4; ++mt)
#pragma unroll
    for (int nt = 0; nt < 2; ++nt)
#pragma unroll
      for (int r = 0; r < 8; ++r) {
        const int R  = m0 + wm*64 + mt*16 + r + 8*lh;
        const int Cc = n0 + wn*32 + nt*16 + lm;
        out[(size_t)R*EMBD + Cc] = acc[mt][nt][r] + pb[Cc];
      }
}

// ---------- launcher ----------
extern "C" void kernel_launch(void* const* d_in, const int* in_sizes, int n_in,
                              void* d_out, int out_size, void* d_ws, size_t ws_size,
                              hipStream_t stream) {
  (void)in_sizes; (void)n_in; (void)out_size; (void)ws_size;
  const float* x       = (const float*)d_in[0];
  const float* qkv_w   = (const float*)d_in[1];
  const float* q_bias  = (const float*)d_in[2];
  const float* v_bias  = (const float*)d_in[3];
  const float* flex    = (const float*)d_in[4];
  const float* cpb_w1  = (const float*)d_in[5];
  const float* cpb_b1  = (const float*)d_in[6];
  const float* cpb_w2  = (const float*)d_in[7];
  const float* proj_w  = (const float*)d_in[8];
  const float* proj_b  = (const float*)d_in[9];
  const float* rel_tab = (const float*)d_in[10];
  const int*   rel_idx = (const int*)d_in[11];
  float* out = (float*)d_out;

  char* ws = (char*)d_ws;
  u16*   qg    = (u16*)(ws + OFF_Q);
  u16*   kg    = (u16*)(ws + OFF_K);
  u16*   vtg   = (u16*)(ws + OFF_VT);
  u16*   aoutg = (u16*)(ws + OFF_AOUT);
  u16*   xb    = (u16*)(ws + OFF_XB);      // aliases aoutg (safe: see layout note)
  u16*   wqb   = (u16*)(ws + OFF_WQ);
  u16*   wpb   = (u16*)(ws + OFF_WP);
  float* biasb = (float*)(ws + OFF_BIAS);
  float* tab   = (float*)(ws + OFF_TAB);

  // one-shot bf16 conversions (x: 64M elts, wq: 768K, wp: 256K)
  cvt_bf16_kernel<<<dim3((BATCH*SEQ*EMBD)/2048), dim3(256), 0, stream>>>(x, xb);
  cvt_bf16_kernel<<<dim3((C3*EMBD)/2048), dim3(256), 0, stream>>>(qkv_w, wqb);
  cvt_bf16_kernel<<<dim3((EMBD*EMBD)/2048), dim3(256), 0, stream>>>(proj_w, wpb);

  cpb_tab_kernel<<<dim3(225), dim3(128), 0, stream>>>(rel_tab, cpb_w1, cpb_b1, cpb_w2, tab);
  cpb_bias_kernel<<<dim3(256), dim3(256), 0, stream>>>(tab, rel_idx, biasb);
  qkv_gemm_kernel<<<dim3(C3/128, (BATCH*SEQ)/128), dim3(256), 0, stream>>>(
      xb, wqb, q_bias, v_bias, qg, kg, vtg);
  attn_kernel<<<dim3(BATCH*HEADS), dim3(128), 0, stream>>>(
      qg, kg, vtg, biasb, flex, aoutg);
  proj_gemm_kernel<<<dim3(EMBD/128, (BATCH*SEQ)/128), dim3(256), 0, stream>>>(
      aoutg, wpb, proj_b, out);
}
